// Model_39676907884736
// MI455X (gfx1250) — matrix operand order repaired and checked by reading, not compiled, not run on hardware
//
#include <hip/hip_runtime.h>

typedef unsigned short u16;
typedef __attribute__((ext_vector_type(16))) __bf16 v16bf;
typedef __attribute__((ext_vector_type(8)))  __bf16 v8bf;
typedef __attribute__((ext_vector_type(8)))  float  v8f;
typedef __attribute__((ext_vector_type(4)))  int    i32x4;

union Frag16 { v16bf v; v8bf h[2]; };

constexpr int CSQ = 2048;
constexpr int CSK = 2048;
constexpr int CD  = 1024;
constexpr int CDH = 64;

#if __has_builtin(__builtin_amdgcn_global_load_async_to_lds_b128) && \
    __has_builtin(__builtin_amdgcn_s_wait_asynccnt)
#define USE_ASYNC_LDS 1
#else
#define USE_ASYNC_LDS 0
#endif

typedef __attribute__((address_space(1))) i32x4 g_i32x4;
typedef __attribute__((address_space(3))) i32x4 l_i32x4;

__device__ __forceinline__ void async_g2l_b128(const u16* g, u16* l) {
#if USE_ASYNC_LDS
  __builtin_amdgcn_global_load_async_to_lds_b128(
      (g_i32x4*)g, (l_i32x4*)l, 0, 0);
#endif
}

__device__ __forceinline__ u16 f2bf(float f) {
  unsigned u = __float_as_uint(f);
  u += 0x7FFFu + ((u >> 16) & 1u);
  return (u16)(u >> 16);
}

// ---------------- fp32 -> bf16 conversion ----------------
__global__ void cvt_kernel(const float* __restrict__ x, u16* __restrict__ y, int n4) {
  int i = blockIdx.x * blockDim.x + threadIdx.x;
  if (i >= n4) return;
  float4 f = ((const float4*)x)[i];
  union { uint2 u; u16 s[4]; } r;
  r.s[0] = f2bf(f.x); r.s[1] = f2bf(f.y); r.s[2] = f2bf(f.z); r.s[3] = f2bf(f.w);
  ((uint2*)y)[i] = r.u;
}

// ---------------- bf16 WMMA GEMM: Y = (X @ W^T + bias) * scale ----------------
// X: (M,K) bf16 row-major, W: (N,K) bf16 row-major. 128x128 block tile,
// 8 waves, each wave computes 32x64 (2x4 WMMA tiles), BK=64,
// double-buffered LDS (64KB) with async global->LDS staging; last iter peeled
// so the steady-state loop has a branch-free s_wait_asynccnt(8).
__global__ __launch_bounds__(256)
void gemm_bf16_wmma(const u16* __restrict__ X, const u16* __restrict__ W,
                    const float* __restrict__ bias,
                    u16* __restrict__ Ybf, float* __restrict__ Yf,
                    int M, int N, int K, float scale)
{
  __shared__ __align__(16) u16 Als[2][128 * 64];
  __shared__ __align__(16) u16 Bls[2][128 * 64];

  const int tid  = threadIdx.x;
  const int lane = tid & 31;
  const int wave = tid >> 5;
  const int hlf  = lane >> 4;     // which 16-lane half
  const int l16  = lane & 15;
  const int waveM = wave & 3;     // 4 row groups of 32
  const int waveN = wave >> 2;    // 2 col groups of 64
  const int m0 = blockIdx.y * 128;
  const int n0 = blockIdx.x * 128;

  v8f acc[2][4];
  #pragma unroll
  for (int i = 0; i < 2; i++)
    #pragma unroll
    for (int j = 0; j < 4; j++)
      #pragma unroll
      for (int e = 0; e < 8; e++) acc[i][j][e] = 0.f;

  // stage a 128x64 A tile + 128x64 B tile: 1024 16B-chunks, 4 per thread each
  auto stage = [&](int buf, int kk) {
#if USE_ASYNC_LDS
    #pragma unroll
    for (int i = 0; i < 4; i++) {
      int c = tid + i * 256;
      int r = c >> 3, kc = (c & 7) << 3;
      async_g2l_b128(&X[(size_t)(m0 + r) * K + kk + kc], &Als[buf][r * 64 + kc]);
      async_g2l_b128(&W[(size_t)(n0 + r) * K + kk + kc], &Bls[buf][r * 64 + kc]);
    }
#else
    uint4 ta[4], tb[4];
    #pragma unroll
    for (int i = 0; i < 4; i++) {
      int c = tid + i * 256;
      int r = c >> 3, kc = (c & 7) << 3;
      ta[i] = *(const uint4*)&X[(size_t)(m0 + r) * K + kk + kc];
      tb[i] = *(const uint4*)&W[(size_t)(n0 + r) * K + kk + kc];
    }
    #pragma unroll
    for (int i = 0; i < 4; i++) {
      int c = tid + i * 256;
      int r = c >> 3, kc = (c & 7) << 3;
      *(uint4*)&Als[buf][r * 64 + kc] = ta[i];
      *(uint4*)&Bls[buf][r * 64 + kc] = tb[i];
    }
#endif
  };

  // 16 WMMAs on one 64-deep LDS tile (two K-substeps of 32)
  auto compute = [&](int buf) {
    #pragma unroll
    for (int ks = 0; ks < 2; ks++) {
      Frag16 a[2], b[4];
      #pragma unroll
      for (int ms = 0; ms < 2; ms++) {
        int r = waveM * 32 + ms * 16 + l16;
        a[ms].h[0] = *(const v8bf*)&Als[buf][r * 64 + ks * 32 + hlf * 8];
        a[ms].h[1] = *(const v8bf*)&Als[buf][r * 64 + ks * 32 + hlf * 8 + 16];
      }
      #pragma unroll
      for (int ns = 0; ns < 4; ns++) {
        int cc = waveN * 64 + ns * 16 + l16;
        b[ns].h[0] = *(const v8bf*)&Bls[buf][cc * 64 + ks * 32 + hlf * 8];
        b[ns].h[1] = *(const v8bf*)&Bls[buf][cc * 64 + ks * 32 + hlf * 8 + 16];
      }
      #pragma unroll
      for (int ms = 0; ms < 2; ms++)
        #pragma unroll
        for (int ns = 0; ns < 4; ns++)
          acc[ms][ns] = __builtin_amdgcn_wmma_f32_16x16x32_bf16(
              false, a[ms].v, false, b[ns].v, (short)0, acc[ms][ns], false, false);
    }
  };

  stage(0, 0);
  const int nk = K >> 6;          // K multiple of 64
  for (int it = 0; it < nk - 1; it++) {
    const int cur = it & 1;
    stage(cur ^ 1, (it + 1) << 6);
#if USE_ASYNC_LDS
    __builtin_amdgcn_s_wait_asynccnt(8);
#endif
    __syncthreads();
    compute(cur);
    __syncthreads();
  }
#if USE_ASYNC_LDS
  __builtin_amdgcn_s_wait_asynccnt(0);
#endif
  __syncthreads();
  compute((nk - 1) & 1);

  #pragma unroll
  for (int ms = 0; ms < 2; ms++) {
    int rbase = m0 + waveM * 32 + ms * 16 + hlf * 8;
    #pragma unroll
    for (int ns = 0; ns < 4; ns++) {
      int gcol = n0 + waveN * 64 + ns * 16 + l16;
      float bv = bias[gcol];
      #pragma unroll
      for (int v = 0; v < 8; v++) {
        float val = (acc[ms][ns][v] + bv) * scale;
        size_t idx = (size_t)(rbase + v) * N + gcol;
        if (Ybf) Ybf[idx] = f2bf(val); else Yf[idx] = val;
      }
    }
  }
}

// ---------------- flash attention (bf16 WMMA, f32 online softmax) ----------------
// Grid: (SQ/128, H, B); block 256 (8 waves). Wave w handles 16 query rows.
__global__ __launch_bounds__(256)
void attn_wmma(const u16* __restrict__ Q, const u16* __restrict__ Kmat,
               const u16* __restrict__ V, const int* __restrict__ mask,
               u16* __restrict__ Out)
{
  __shared__ __align__(16) u16 Kls[64 * 64];        // [key][dh]
  __shared__ __align__(16) u16 Vls[64 * 64];        // transposed: [dh][key]
  __shared__ __align__(16) u16 Pls[8 * 16 * 64];    // per-wave P scratch

  const int tid  = threadIdx.x;
  const int lane = tid & 31;
  const int wave = tid >> 5;
  const int hlf  = lane >> 4;
  const int l16  = lane & 15;
  const int b = blockIdx.z, h = blockIdx.y;
  const int q0 = blockIdx.x * 128 + wave * 16;

  // Q fragments: 16 rows x 64 dh -> two A-frags (K-steps of 32)
  Frag16 qf[2];
  const u16* qrow = Q + ((size_t)b * CSQ + q0 + l16) * CD + h * CDH;
  #pragma unroll
  for (int s = 0; s < 2; s++) {
    qf[s].h[0] = *(const v8bf*)(qrow + s * 32 + hlf * 8);
    qf[s].h[1] = *(const v8bf*)(qrow + s * 32 + hlf * 8 + 16);
  }

  float mrow[8], lrow[8];
  v8f o[4];
  #pragma unroll
  for (int v = 0; v < 8; v++) { mrow[v] = -3.0e38f; lrow[v] = 0.f; }
  #pragma unroll
  for (int d = 0; d < 4; d++)
    #pragma unroll
    for (int e = 0; e < 8; e++) o[d][e] = 0.f;

  u16* pw = &Pls[wave * 16 * 64];

  // this thread's staging chunks (8 bf16 each), 512 chunks per 64x64 tile
  const int kr0 = tid >> 3,         dc0 = (tid & 7) << 3;
  const int kr1 = (tid + 256) >> 3, dc1 = ((tid + 256) & 7) << 3;

  for (int kc = 0; kc < CSK; kc += 64) {
    const u16* kbase = Kmat + ((size_t)b * CSK + kc) * CD + h * CDH;
    const u16* vbase = V    + ((size_t)b * CSK + kc) * CD + h * CDH;
#if USE_ASYNC_LDS
    async_g2l_b128(kbase + (size_t)kr0 * CD + dc0, &Kls[kr0 * 64 + dc0]);
    async_g2l_b128(kbase + (size_t)kr1 * CD + dc1, &Kls[kr1 * 64 + dc1]);
#else
    uint4 ka = *(const uint4*)(kbase + (size_t)kr0 * CD + dc0);
    uint4 kb = *(const uint4*)(kbase + (size_t)kr1 * CD + dc1);
#endif
    union { uint4 u; u16 s[8]; } v0, v1;
    v0.u = *(const uint4*)(vbase + (size_t)kr0 * CD + dc0);
    v1.u = *(const uint4*)(vbase + (size_t)kr1 * CD + dc1);
#if !USE_ASYNC_LDS
    *(uint4*)&Kls[kr0 * 64 + dc0] = ka;
    *(uint4*)&Kls[kr1 * 64 + dc1] = kb;
#endif
    #pragma unroll
    for (int j = 0; j < 8; j++) {
      Vls[(dc0 + j) * 64 + kr0] = v0.s[j];
      Vls[(dc1 + j) * 64 + kr1] = v1.s[j];
    }
#if USE_ASYNC_LDS
    __builtin_amdgcn_s_wait_asynccnt(0);
#endif
    __syncthreads();

    // S = Q @ K^T  (4 tiles of 16 keys, contraction over dh=64 in two steps)
    v8f S[4];
    #pragma unroll
    for (int ct = 0; ct < 4; ct++) {
      Frag16 bk0, bk1;
      int krow = ct * 16 + l16;
      bk0.h[0] = *(const v8bf*)&Kls[krow * 64 + hlf * 8];
      bk0.h[1] = *(const v8bf*)&Kls[krow * 64 + hlf * 8 + 16];
      bk1.h[0] = *(const v8bf*)&Kls[krow * 64 + 32 + hlf * 8];
      bk1.h[1] = *(const v8bf*)&Kls[krow * 64 + 32 + hlf * 8 + 16];
      v8f c;
      #pragma unroll
      for (int e = 0; e < 8; e++) c[e] = 0.f;
      c = __builtin_amdgcn_wmma_f32_16x16x32_bf16(false, qf[0].v, false, bk0.v,
                                                  (short)0, c, false, false);
      c = __builtin_amdgcn_wmma_f32_16x16x32_bf16(false, qf[1].v, false, bk1.v,
                                                  (short)0, c, false, false);
      float bv = ((float)mask[b * CSK + kc + ct * 16 + l16] - 1.f) * 1.0e9f;
      #pragma unroll
      for (int e = 0; e < 8; e++) c[e] += bv;
      S[ct] = c;
    }

    // online softmax: rows live in 16-lane halves, reduce with shfl_xor
    #pragma unroll
    for (int v = 0; v < 8; v++) {
      float t = fmaxf(fmaxf(S[0][v], S[1][v]), fmaxf(S[2][v], S[3][v]));
      #pragma unroll
      for (int off = 8; off >= 1; off >>= 1)
        t = fmaxf(t, __shfl_xor(t, off, 32));
      float nm = fmaxf(mrow[v], t);
      float alpha = __expf(mrow[v] - nm);
      mrow[v] = nm;
      lrow[v] *= alpha;
      #pragma unroll
      for (int d = 0; d < 4; d++) o[d][v] *= alpha;
    }

    // P = exp(S - m): accumulate row sums, spill to per-wave LDS as bf16
    #pragma unroll
    for (int v = 0; v < 8; v++) {
      int r = v + hlf * 8;
      float rs = 0.f;
      #pragma unroll
      for (int ct = 0; ct < 4; ct++) {
        float p = __expf(S[ct][v] - mrow[v]);
        rs += p;
        pw[r * 64 + ct * 16 + l16] = f2bf(p);
      }
      #pragma unroll
      for (int off = 8; off >= 1; off >>= 1)
        rs += __shfl_xor(rs, off, 32);
      lrow[v] += rs;
    }

    // O += P @ V   (A-frags re-read from LDS; B-frags from transposed V)
    Frag16 pf[2];
    #pragma unroll
    for (int s = 0; s < 2; s++) {
      pf[s].h[0] = *(const v8bf*)&pw[l16 * 64 + s * 32 + hlf * 8];
      pf[s].h[1] = *(const v8bf*)&pw[l16 * 64 + s * 32 + hlf * 8 + 16];
    }
    #pragma unroll
    for (int d = 0; d < 4; d++) {
      Frag16 bv0, bv1;
      int dr = d * 16 + l16;
      bv0.h[0] = *(const v8bf*)&Vls[dr * 64 + hlf * 8];
      bv0.h[1] = *(const v8bf*)&Vls[dr * 64 + hlf * 8 + 16];
      bv1.h[0] = *(const v8bf*)&Vls[dr * 64 + 32 + hlf * 8];
      bv1.h[1] = *(const v8bf*)&Vls[dr * 64 + 32 + hlf * 8 + 16];
      o[d] = __builtin_amdgcn_wmma_f32_16x16x32_bf16(false, pf[0].v, false, bv0.v,
                                                     (short)0, o[d], false, false);
      o[d] = __builtin_amdgcn_wmma_f32_16x16x32_bf16(false, pf[1].v, false, bv1.v,
                                                     (short)0, o[d], false, false);
    }
    __syncthreads();
  }

  // normalize and write merged-head output (B, SQ, D) bf16
  #pragma unroll
  for (int v = 0; v < 8; v++) {
    float inv = 1.f / lrow[v];
    size_t row = (size_t)b * CSQ + q0 + v + hlf * 8;
    #pragma unroll
    for (int d = 0; d < 4; d++)
      Out[row * CD + h * CDH + d * 16 + l16] = f2bf(o[d][v] * inv);
  }
}

extern "C" void kernel_launch(void* const* d_in, const int* in_sizes, int n_in,
                              void* d_out, int out_size, void* d_ws, size_t ws_size,
                              hipStream_t stream) {
  const float* q      = (const float*)d_in[0];
  const float* k      = (const float*)d_in[1];
  const float* v      = (const float*)d_in[2];
  const float* Win_b  = (const float*)d_in[4];
  const float* Wff_b  = (const float*)d_in[6];
  const float* Wout_b = (const float*)d_in[8];
  const int*   mask   = (const int*)d_in[9];
  float* out = (float*)d_out;

  u16* ws = (u16*)d_ws;
  const size_t WE = (size_t)1024 * 1024;   // weight elems
  const size_t TE = (size_t)4096 * 1024;   // activation elems (B*S*D)
  u16* WinB  = ws;
  u16* WffB  = WinB + WE;
  u16* WoutB = WffB + WE;
  u16* qB   = WoutB + WE;
  u16* kB   = qB + TE;
  u16* vB   = kB + TE;
  u16* qhB  = vB + TE;
  u16* khB  = qhB + TE;
  u16* vhB  = khB + TE;
  u16* tB   = vhB + TE;
  u16* aoB  = tB + TE;
  u16* fB   = aoB + TE;

  auto cvt = [&](const float* x, u16* y, size_t n) {
    int n4 = (int)(n / 4);
    cvt_kernel<<<(n4 + 255) / 256, 256, 0, stream>>>(x, y, n4);
  };
  cvt((const float*)d_in[3], WinB, WE);
  cvt((const float*)d_in[5], WffB, WE);
  cvt((const float*)d_in[7], WoutB, WE);
  cvt(q, qB, TE);
  cvt(k, kB, TE);
  cvt(v, vB, TE);

  dim3 gg(1024 / 128, 4096 / 128);
  const float rsD = 0.03125f;  // D^-0.5, D=1024
  // qh = (q @ Win^T + b) * D^-0.5
  gemm_bf16_wmma<<<gg, 256, 0, stream>>>(qB, WinB, Win_b, qhB, nullptr, 4096, 1024, 1024, rsD);
  // kh = (k @ Win^T + b) @ Wout^T + b
  gemm_bf16_wmma<<<gg, 256, 0, stream>>>(kB, WinB, Win_b, tB, nullptr, 4096, 1024, 1024, 1.f);
  gemm_bf16_wmma<<<gg, 256, 0, stream>>>(tB, WoutB, Wout_b, khB, nullptr, 4096, 1024, 1024, 1.f);
  // vh = (v @ Win^T + b) @ Wout^T + b
  gemm_bf16_wmma<<<gg, 256, 0, stream>>>(vB, WinB, Win_b, tB, nullptr, 4096, 1024, 1024, 1.f);
  gemm_bf16_wmma<<<gg, 256, 0, stream>>>(tB, WoutB, Wout_b, vhB, nullptr, 4096, 1024, 1024, 1.f);

  dim3 ga(2048 / 128, 16, 2);
  attn_wmma<<<ga, 256, 0, stream>>>(qhB, khB, vhB, mask, aoB);

  // out = (ao @ Wff^T + b) @ Wout^T + b   (final GEMM writes fp32 to d_out)
  gemm_bf16_wmma<<<gg, 256, 0, stream>>>(aoB, WffB, Wff_b, fB, nullptr, 4096, 1024, 1024, 1.f);
  gemm_bf16_wmma<<<gg, 256, 0, stream>>>(fB, WoutB, Wout_b, nullptr, out, 4096, 1024, 1024, 1.f);
}
